// Council_58050777972841
// MI455X (gfx1250) — matrix operand order repaired and checked by reading, not compiled, not run on hardware
//
#include <hip/hip_runtime.h>

// Problem constants (match reference): D is [B, N, N] f32, B=256, N=512.
#define N       512
#define MAXIT   100
#define EPSF    1e-6f

typedef __attribute__((ext_vector_type(16))) _Float16 v16h;
typedef __attribute__((ext_vector_type(8)))  float    v8f;

// ---------------------------------------------------------------------------
// Kernel 1: per-row stats. One wave per row: outgoing = rowsum - diag + eps.
// Stores diag[row] and 1/outgoing[row].
// ---------------------------------------------------------------------------
__global__ __launch_bounds__(256) void council_rowstats(
    const float* __restrict__ D, float* __restrict__ diag,
    float* __restrict__ invout)
{
    const int tid  = threadIdx.x;
    const int lane = tid & 31;
    const int wid  = tid >> 5;
    const int row  = blockIdx.x * 8 + wid;          // row = b*N + i
    const int i    = row & (N - 1);

    const float* p = D + (size_t)row * N;
    float s = 0.0f;
    #pragma unroll
    for (int t = 0; t < N / 32; ++t) s += p[lane + 32 * t];
    // wave32 reduction
    #pragma unroll
    for (int off = 16; off > 0; off >>= 1) s += __shfl_xor(s, off, 32);

    if (lane == 0) {
        const float dg = p[i];
        const float og = s - dg + EPSF;
        diag[row]   = dg;
        invout[row] = 1.0f / og;
    }
}

// ---------------------------------------------------------------------------
// Kernel 2: normalize, zero diagonal, convert to f16, store TRANSPOSED:
//   Ttr[b][j][i] = (i==j ? 0 : D[b][i][j]) / outgoing[b][i]
// 32x32 tiles through LDS so both global read and write are coalesced.
// ---------------------------------------------------------------------------
__global__ __launch_bounds__(256) void council_normT(
    const float* __restrict__ D, const float* __restrict__ invout,
    _Float16* __restrict__ Ttr)
{
    const int b    = blockIdx.x >> 8;               // 256 tiles per batch
    const int tile = blockIdx.x & 255;
    const int ti   = (tile >> 4) << 5;              // i-tile base
    const int tj   = (tile & 15) << 5;              // j-tile base
    const int tx   = threadIdx.x & 31;
    const int ty   = threadIdx.x >> 5;              // 0..7

    __shared__ float tl[32][33];

    #pragma unroll
    for (int r = 0; r < 4; ++r) {
        const int iy = ty + r * 8;
        const int i  = ti + iy;
        const int j  = tj + tx;
        float v = D[((size_t)b * N + i) * N + j] * invout[b * N + i];
        if (i == j) v = 0.0f;
        tl[iy][tx] = v;
    }
    __syncthreads();
    #pragma unroll
    for (int r = 0; r < 4; ++r) {
        const int jy = ty + r * 8;
        const int j  = tj + jy;
        const int i  = ti + tx;
        Ttr[((size_t)b * N + j) * N + i] = (_Float16)tl[tx][jy];
    }
}

// ---------------------------------------------------------------------------
// Kernel 3: the 100-step power iteration. One workgroup (8 waves) per batch.
// Wave w owns output columns [w*64, w*64+64) as 4 WMMA j-tiles.
//   A (16x32 f16): outflow broadcast across M  -> built from LDS
//   B (32x16 f16): Ttr columns, contiguous 32B per lane
//   D row 0 (VGPR0, lanes 0..15) = new_del for the j-tile
// ---------------------------------------------------------------------------
__global__ __launch_bounds__(256) void council_iterate(
    const _Float16* __restrict__ Ttr, const float* __restrict__ diag,
    float* __restrict__ out)
{
    const int b    = blockIdx.x;
    const int tid  = threadIdx.x;
    const int lane = tid & 31;
    const int wid  = tid >> 5;
    const int col  = lane & 15;      // N index within a 16-wide tile
    const int hi   = lane >> 4;      // half-wave selector for WMMA layouts
    const int jb0  = wid * 64;       // first of this wave's 4 j-tiles

    __shared__ _Float16 of_h[N];     // outflow (f16) for A-matrix
    __shared__ float    nd[N];       // new_del

    // Each thread owns rows i0 = tid and i1 = tid+256 of this batch.
    const int i0 = tid, i1 = tid + 256;
    const float d0 = diag[b * N + i0];
    const float d1 = diag[b * N + i1];
    float pd0 = 1.0f, pd1 = 1.0f, pk0 = 0.0f, pk1 = 0.0f;

    const _Float16* __restrict__ Tb = Ttr + (size_t)b * N * N;

    for (int it = 0; it < MAXIT; ++it) {
        // --- elementwise phase: keep / outflow -------------------------------
        const float k0 = d0 * pd0, k1 = d1 * pd1;
        pk0 += k0; pk1 += k1;
        of_h[i0] = (_Float16)(pd0 - k0);
        of_h[i1] = (_Float16)(pd1 - k1);
        __syncthreads();

        // --- WMMA phase: new_del[j] = sum_i outflow[i] * T[i][j] -------------
        v8f c0 = {}, c1 = {}, c2 = {}, c3 = {};
        #pragma unroll 4
        for (int kb = 0; kb < N; kb += 32) {
            // A fragment: lanes 0-15 hold K {0..15}, lanes 16-31 {16..31}
            v16h a;
            const int ka = kb + hi * 16;
            #pragma unroll
            for (int t = 0; t < 8; ++t) {
                a[t]     = of_h[ka + t];
                a[t + 8] = of_h[ka + 8 + t];
            }
            // B fragments: lane holds 16 consecutive K of column (jb + col)
            const _Float16* brow =
                Tb + (size_t)(jb0 + col) * N + kb + hi * 16;
            const v16h bm0 = *(const v16h*)(brow);
            const v16h bm1 = *(const v16h*)(brow + 16 * N);
            const v16h bm2 = *(const v16h*)(brow + 32 * N);
            const v16h bm3 = *(const v16h*)(brow + 48 * N);
            c0 = __builtin_amdgcn_wmma_f32_16x16x32_f16(false, a, false, bm0,
                                                        (short)0, c0, false, false);
            c1 = __builtin_amdgcn_wmma_f32_16x16x32_f16(false, a, false, bm1,
                                                        (short)0, c1, false, false);
            c2 = __builtin_amdgcn_wmma_f32_16x16x32_f16(false, a, false, bm2,
                                                        (short)0, c2, false, false);
            c3 = __builtin_amdgcn_wmma_f32_16x16x32_f16(false, a, false, bm3,
                                                        (short)0, c3, false, false);
        }
        // D-matrix row M=0 lives in VGPR0 of lanes 0..15 (N = lane)
        if (lane < 16) {
            nd[jb0 +      lane] = c0[0];
            nd[jb0 + 16 + lane] = c1[0];
            nd[jb0 + 32 + lane] = c2[0];
            nd[jb0 + 48 + lane] = c3[0];
        }
        __syncthreads();
        pd0 = nd[i0];
        pd1 = nd[i1];
    }

    out[b * N + i0] = pd0 + pk0;
    out[b * N + i1] = pd1 + pk1;
}

// ---------------------------------------------------------------------------
// Launch. Workspace layout (needs ~129 MB):
//   [0, B*N*N*2)                      : Ttr (f16)
//   [+0, B*N*4)                       : diag (f32)
//   [+B*N*4, +2*B*N*4)                : invout (f32)
// ---------------------------------------------------------------------------
extern "C" void kernel_launch(void* const* d_in, const int* in_sizes, int n_in,
                              void* d_out, int out_size, void* d_ws, size_t ws_size,
                              hipStream_t stream) {
    const float* D = (const float*)d_in[0];
    const int B = in_sizes[0] / (N * N);           // 256

    char* ws = (char*)d_ws;
    _Float16* Ttr   = (_Float16*)ws;
    float*    diag  = (float*)(ws + (size_t)B * N * N * sizeof(_Float16));
    float*    invout = diag + (size_t)B * N;
    float*    out   = (float*)d_out;

    council_rowstats<<<B * N / 8, 256, 0, stream>>>(D, diag, invout);
    council_normT  <<<B * 256,   256, 0, stream>>>(D, invout, Ttr);
    council_iterate<<<B,         256, 0, stream>>>(Ttr, diag, out);
}
